// ASTGCN_Block_80410377715727
// MI455X (gfx1250) — matrix operand order repaired and checked by reading, not compiled, not run on hardware
//
#include <hip/hip_runtime.h>
#include <hip/hip_bf16.h>

// ---------------- problem constants ----------------
#define B_  8
#define N_  2048
#define F_  16
#define T_  24
#define K_  3
#define C_  64
#define TF_ 64
#define FT_ (F_ * T_)   // 384
#define LN_EPS 1e-5f

typedef __attribute__((ext_vector_type(16))) __bf16 bf16x16;
typedef __attribute__((ext_vector_type(8)))  __bf16 bf16x8;
typedef __attribute__((ext_vector_type(8)))  float  f32x8;

// GEMM tiling: 128x128 block tile, 4 waves, each wave 64x64 (4x4 WMMA subtiles)
#define BM 128
#define BN 128
#define BK 32
#define LDA 40   // elements; 40*2B = 80B rows -> 16B aligned
#define LDB 40

// ============================================================
// K1: inner[b,t,f] = sum_n x[b,n,f,t] * U1[n]   (block reduction)
// ============================================================
__global__ __launch_bounds__(256) void k_inner(const float* __restrict__ x,
                                               const float* __restrict__ U1,
                                               float* __restrict__ inner) {
  int bi = blockIdx.x;                 // over B*F*T
  int b = bi / (F_ * T_);
  int rem = bi % (F_ * T_);
  int f = rem / T_;
  int t = rem % T_;
  float s = 0.f;
  for (int n = threadIdx.x; n < N_; n += 256)
    s += x[(((size_t)b * N_ + n) * F_ + f) * T_ + t] * U1[n];
  __shared__ float red[256];
  red[threadIdx.x] = s;
  __syncthreads();
  for (int off = 128; off > 0; off >>= 1) {
    if (threadIdx.x < off) red[threadIdx.x] += red[threadIdx.x + off];
    __syncthreads();
  }
  if (threadIdx.x == 0) inner[(b * T_ + t) * F_ + f] = red[0];
}

// ============================================================
// K2: rhsT[b,n,u] = sum_f U3[f] * x[b,n,f,u]
// ============================================================
__global__ __launch_bounds__(256) void k_rhsT(const float* __restrict__ x,
                                              const float* __restrict__ U3,
                                              float* __restrict__ rhsT) {
  size_t idx = (size_t)blockIdx.x * 256 + threadIdx.x;
  if (idx >= (size_t)B_ * N_ * T_) return;
  int u = idx % T_;
  size_t bn = idx / T_;   // b*N+n
  float s = 0.f;
  for (int f = 0; f < F_; ++f)
    s += U3[f] * x[(bn * F_ + f) * T_ + u];
  rhsT[idx] = s;
}

// ============================================================
// K3: per-batch TAT: M1 = U2 @ rhsT ; prod = inner @ M1 ;
//     E = Ve @ sigmoid(prod+be) ; tat = softmax(E, axis=rows)
// ============================================================
__global__ __launch_bounds__(256) void k_tat(const float* __restrict__ inner,
                                             const float* __restrict__ rhsT,
                                             const float* __restrict__ U2,
                                             const float* __restrict__ Ve,
                                             const float* __restrict__ be,
                                             float* __restrict__ tat) {
  int b = blockIdx.x;
  int tid = threadIdx.x;
  __shared__ float M1[F_ * T_];    // [f][u]
  __shared__ float innS[T_ * F_];  // [t][f]
  __shared__ float ps[T_ * T_];    // sigmoid(prod+be) [w][u]
  __shared__ float Es[T_ * T_];
  for (int e = tid; e < T_ * F_; e += 256) innS[e] = inner[b * T_ * F_ + e];
  for (int e = tid; e < F_ * T_; e += 256) {
    int f = e / T_, u = e % T_;
    float s = 0.f;
    for (int n = 0; n < N_; ++n)
      s += U2[(size_t)f * N_ + n] * rhsT[((size_t)b * N_ + n) * T_ + u];
    M1[e] = s;
  }
  __syncthreads();
  for (int e = tid; e < T_ * T_; e += 256) {
    int w = e / T_, u = e % T_;
    float s = 0.f;
    for (int f = 0; f < F_; ++f) s += innS[w * F_ + f] * M1[f * T_ + u];
    s += be[w * T_ + u];
    ps[e] = 1.f / (1.f + __expf(-s));
  }
  __syncthreads();
  for (int e = tid; e < T_ * T_; e += 256) {
    int t = e / T_, u = e % T_;
    float s = 0.f;
    for (int w = 0; w < T_; ++w) s += Ve[t * T_ + w] * ps[w * T_ + u];
    Es[e] = s;
  }
  __syncthreads();
  if (tid < T_) {        // softmax over rows t, one column u per thread
    int u = tid;
    float mx = -1e30f;
    for (int t = 0; t < T_; ++t) mx = fmaxf(mx, Es[t * T_ + u]);
    float sm = 0.f;
    for (int t = 0; t < T_; ++t) sm += __expf(Es[t * T_ + u] - mx);
    float inv = 1.f / sm;
    for (int t = 0; t < T_; ++t)
      tat[(b * T_ + t) * T_ + u] = __expf(Es[t * T_ + u] - mx) * inv;
  }
}

// ============================================================
// K4: x_tat[b,n,f,u] = sum_t x[b,n,f,t] * tat[b,t,u]
// ============================================================
__global__ __launch_bounds__(256) void k_xtat(const float* __restrict__ x,
                                              const float* __restrict__ tat,
                                              float* __restrict__ x_tat) {
  int b = blockIdx.y;
  __shared__ float tt[T_ * T_];
  for (int e = threadIdx.x; e < T_ * T_; e += 256) tt[e] = tat[b * T_ * T_ + e];
  __syncthreads();
  int nf = blockIdx.x * 256 + threadIdx.x;   // 0 .. N*F-1
  const float* xr = &x[((size_t)b * N_ * F_ + nf) * T_];
  float row[T_];
  for (int t = 0; t < T_; ++t) row[t] = xr[t];
  float* o = &x_tat[((size_t)b * N_ * F_ + nf) * T_];
  for (int u = 0; u < T_; ++u) {
    float s = 0.f;
    for (int t = 0; t < T_; ++t) s += row[t] * tt[t * T_ + u];
    o[u] = s;
  }
}

// ============================================================
// K5: elementwise f32 -> bf16
// ============================================================
__global__ __launch_bounds__(256) void k_tobf(const float* __restrict__ src,
                                              __bf16* __restrict__ dst,
                                              size_t count) {
  size_t i = (size_t)blockIdx.x * 256 + threadIdx.x;
  if (i < count) dst[i] = (__bf16)src[i];
}

// ============================================================
// K6: SAT prep: lhs2[b,n,t], rhs2v[b,n,t] from x_tat
// ============================================================
__global__ __launch_bounds__(256) void k_satprep(const float* __restrict__ x_tat,
                                                 const float* __restrict__ W1,
                                                 const float* __restrict__ W2,
                                                 const float* __restrict__ W3,
                                                 float* __restrict__ lhs2,
                                                 float* __restrict__ rhs2v) {
  int idx = blockIdx.x * 256 + threadIdx.x;
  if (idx >= B_ * N_) return;
  const float* xr = &x_tat[(size_t)idx * F_ * T_];
  float v[F_];
  for (int f = 0; f < F_; ++f) {
    float s = 0.f;
    for (int t = 0; t < T_; ++t) s += xr[f * T_ + t] * W1[t];
    v[f] = s;
  }
  for (int t = 0; t < T_; ++t) {
    float s = 0.f;
    for (int f = 0; f < F_; ++f) s += v[f] * W2[f * T_ + t];
    lhs2[(size_t)idx * T_ + t] = s;
  }
  for (int t = 0; t < T_; ++t) {
    float s = 0.f;
    for (int f = 0; f < F_; ++f) s += W3[f] * xr[f * T_ + t];
    rhs2v[(size_t)idx * T_ + t] = s;
  }
}

// ============================================================
// K7: P[b,n,m] = sigmoid( dot_t(lhs2[b,n,:], rhs2v[b,m,:]) + bs[n,m] ) -> bf16
// ============================================================
__global__ __launch_bounds__(256) void k_prod2(const float* __restrict__ lhs2,
                                               const float* __restrict__ rhs2v,
                                               const float* __restrict__ bs,
                                               __bf16* __restrict__ P) {
  int b = blockIdx.z, n = blockIdx.y, m0 = blockIdx.x * 256;
  __shared__ float lrow[T_];
  __shared__ float rt[256 * T_];
  if (threadIdx.x < T_) lrow[threadIdx.x] = lhs2[((size_t)b * N_ + n) * T_ + threadIdx.x];
  for (int e = threadIdx.x; e < 256 * T_; e += 256) {
    int j = e / T_, t = e % T_;
    rt[e] = rhs2v[((size_t)b * N_ + m0 + j) * T_ + t];
  }
  __syncthreads();
  int m = m0 + threadIdx.x;
  float s = 0.f;
  for (int t = 0; t < T_; ++t) s += lrow[t] * rt[threadIdx.x * T_ + t];
  s += bs[(size_t)n * N_ + m];
  P[(size_t)b * N_ * N_ + (size_t)n * N_ + m] = (__bf16)(1.f / (1.f + __expf(-s)));
}

// ============================================================
// WMMA helpers: fragment assembly from LDS via aligned b128 reads.
// A-frag (16x32): lane half h reads K runs {8h..8h+7} and {16+8h..16+8h+7}.
// B-frag (32x16), Bs stored [n][k]: lane half h reads K runs {8h..8h+7} and {16+8h..16+8h+7}.
// (CDNA5 ISA 7.12.2, wave32)
// ============================================================
__device__ __forceinline__ bf16x16 frag_from_runs(const __bf16* base, int off0, int off1) {
  bf16x8 lo = *(const bf16x8*)(base + off0);
  bf16x8 hi = *(const bf16x8*)(base + off1);
  bf16x16 v;
#pragma unroll
  for (int j = 0; j < 8; ++j) { v[j] = lo[j]; v[j + 8] = hi[j]; }
  return v;
}

__device__ __forceinline__ void mma_128x128(const __bf16* As, const __bf16* Bs,
                                            f32x8 acc[4][4]) {
  const int lane = threadIdx.x & 31;
  const int wave = threadIdx.x >> 5;
  const int wm = (wave >> 1) & 1, wn = wave & 1;
  const int r = lane & 15, h = lane >> 4;
  bf16x16 a[4], bfr[4];
#pragma unroll
  for (int s = 0; s < 4; ++s) {
    int row = wm * 64 + s * 16 + r;
    a[s] = frag_from_runs(As + row * LDA, 8 * h, 16 + 8 * h);
    int col = wn * 64 + s * 16 + r;
    bfr[s] = frag_from_runs(Bs + col * LDB, 8 * h, 16 + 8 * h);
  }
#pragma unroll
  for (int i = 0; i < 4; ++i)
#pragma unroll
    for (int j = 0; j < 4; ++j)
      acc[i][j] = __builtin_amdgcn_wmma_f32_16x16x32_bf16(
          false, a[i], false, bfr[j], (short)0, acc[i][j], false, false);
}

// ============================================================
// K8: GEMM1  S[b] = Vs(bf16) @ P[b](bf16), f32 out.  M=N=K=2048
//     128x128 block tile, register double-buffered staging.
// ============================================================
__global__ __launch_bounds__(128) void k_gemm1(const __bf16* __restrict__ Vsbf,
                                               const __bf16* __restrict__ P,
                                               float* __restrict__ S) {
  const int m0 = blockIdx.x * BM, n0 = blockIdx.y * BN, b = blockIdx.z;
  const int tid = threadIdx.x;
  __shared__ __attribute__((aligned(16))) __bf16 As[BM * LDA];
  __shared__ __attribute__((aligned(16))) __bf16 Bs[BN * LDB];
  f32x8 acc[4][4];
#pragma unroll
  for (int i = 0; i < 4; ++i)
#pragma unroll
    for (int j = 0; j < 4; ++j)
#pragma unroll
      for (int e = 0; e < 8; ++e) acc[i][j][e] = 0.f;

  bf16x8 aReg[4], bReg[4];
  auto loadA = [&](int k0) {
#pragma unroll
    for (int i2 = 0; i2 < 4; ++i2) {
      int c = tid + 128 * i2;
      int ml = c >> 2, kc = (c & 3) * 8;   // 128 rows x 4 chunks of 8 k
      aReg[i2] = *(const bf16x8*)(Vsbf + (size_t)(m0 + ml) * N_ + k0 + kc);
    }
  };
  auto loadB = [&](int k0) {
#pragma unroll
    for (int i2 = 0; i2 < 4; ++i2) {
      int c = tid + 128 * i2;
      int ki = c >> 4, nc = (c & 15) * 8;  // 32 k-rows x 16 chunks of 8 n
      bReg[i2] = *(const bf16x8*)(P + (size_t)b * N_ * N_ +
                                  (size_t)(k0 + ki) * N_ + n0 + nc);
    }
  };
  loadA(0);
  loadB(0);
  for (int k0 = 0; k0 < N_; k0 += BK) {
#pragma unroll
    for (int i2 = 0; i2 < 4; ++i2) {       // A: straight vector store
      int c = tid + 128 * i2;
      int ml = c >> 2, kc = (c & 3) * 8;
      *(bf16x8*)(As + ml * LDA + kc) = aReg[i2];
    }
#pragma unroll
    for (int i2 = 0; i2 < 4; ++i2) {       // B: transpose-on-store -> [n][k]
      int c = tid + 128 * i2;
      int ki = c >> 4, nc = (c & 15) * 8;
#pragma unroll
      for (int e = 0; e < 8; ++e) Bs[(nc + e) * LDB + ki] = bReg[i2][e];
    }
    __syncthreads();
    if (k0 + BK < N_) {                    // prefetch next tile into regs
      loadA(k0 + BK);
      loadB(k0 + BK);
      if (k0 + 2 * BK < N_) {
        __builtin_prefetch(Vsbf + (size_t)(m0 + (tid >> 2)) * N_ + k0 + 2 * BK, 0, 3);
        __builtin_prefetch(P + (size_t)b * N_ * N_ +
                           (size_t)(k0 + 2 * BK + (tid >> 4)) * N_ + n0, 0, 3);
      }
    }
    mma_128x128(As, Bs, acc);
    __syncthreads();
  }
  const int lane = tid & 31, wave = tid >> 5;
  const int wm = (wave >> 1) & 1, wn = wave & 1, r = lane & 15, h = lane >> 4;
#pragma unroll
  for (int s0 = 0; s0 < 4; ++s0)
#pragma unroll
    for (int s1 = 0; s1 < 4; ++s1)
#pragma unroll
      for (int i = 0; i < 8; ++i) {
        int m = m0 + wm * 64 + s0 * 16 + 8 * h + i;
        int n = n0 + wn * 64 + s1 * 16 + r;
        S[(size_t)b * N_ * N_ + (size_t)m * N_ + n] = acc[s0][s1][i];
      }
}

// ============================================================
// K9: column softmax over axis=1 of S[b,i,m] -> sat bf16
// ============================================================
__global__ __launch_bounds__(64) void k_satsm(const float* __restrict__ S,
                                              __bf16* __restrict__ satbf) {
  int b = blockIdx.y;
  int col = blockIdx.x * 64 + threadIdx.x;
  const float* Sc = &S[(size_t)b * N_ * N_ + col];
  float mx = -1e30f;
  for (int i = 0; i < N_; ++i) mx = fmaxf(mx, Sc[(size_t)i * N_]);
  float sm = 0.f;
  for (int i = 0; i < N_; ++i) sm += __expf(Sc[(size_t)i * N_] - mx);
  float inv = 1.f / sm;
  __bf16* oc = &satbf[(size_t)b * N_ * N_ + col];
  for (int i = 0; i < N_; ++i)
    oc[(size_t)i * N_] = (__bf16)(__expf(Sc[(size_t)i * N_] - mx) * inv);
}

// ============================================================
// K10: GEMM2  r_k[b,m,ft] = sum_q (cheb[k,q,m]*sat[b,q,m]) * xbf[b,q,ft]
//      gating fused into A-tile staging (transpose-on-store).
//      M=2048, Nout=384 (=3*128), K=2048
// ============================================================
__global__ __launch_bounds__(128) void k_gemm2(const float* __restrict__ cheb,
                                               const __bf16* __restrict__ satbf,
                                               const __bf16* __restrict__ xbf,
                                               float* __restrict__ rbuf) {
  const int m0 = blockIdx.x * BM, n0 = blockIdx.y * BN;
  const int b = blockIdx.z / K_, kk = blockIdx.z % K_;
  const int tid = threadIdx.x;
  __shared__ __attribute__((aligned(16))) __bf16 As[BM * LDA];
  __shared__ __attribute__((aligned(16))) __bf16 Bs[BN * LDB];
  f32x8 acc[4][4];
#pragma unroll
  for (int i = 0; i < 4; ++i)
#pragma unroll
    for (int j = 0; j < 4; ++j)
#pragma unroll
      for (int e = 0; e < 8; ++e) acc[i][j][e] = 0.f;

  bf16x8 aReg[4], bReg[4];
  auto loadA = [&](int q0) {
#pragma unroll
    for (int i2 = 0; i2 < 4; ++i2) {
      int c = tid + 128 * i2;
      int qi = c >> 4, mc = (c & 15) * 8;  // 32 q-rows x 16 chunks of 8 m
      const float* cp = cheb + ((size_t)kk * N_ + q0 + qi) * N_ + m0 + mc;
      float4 c0 = *(const float4*)(cp);
      float4 c1 = *(const float4*)(cp + 4);
      bf16x8 sv = *(const bf16x8*)(satbf + (size_t)b * N_ * N_ +
                                   (size_t)(q0 + qi) * N_ + m0 + mc);
      bf16x8 o;
      o[0] = (__bf16)(c0.x * (float)sv[0]);
      o[1] = (__bf16)(c0.y * (float)sv[1]);
      o[2] = (__bf16)(c0.z * (float)sv[2]);
      o[3] = (__bf16)(c0.w * (float)sv[3]);
      o[4] = (__bf16)(c1.x * (float)sv[4]);
      o[5] = (__bf16)(c1.y * (float)sv[5]);
      o[6] = (__bf16)(c1.z * (float)sv[6]);
      o[7] = (__bf16)(c1.w * (float)sv[7]);
      aReg[i2] = o;
    }
  };
  auto loadB = [&](int q0) {
#pragma unroll
    for (int i2 = 0; i2 < 4; ++i2) {
      int c = tid + 128 * i2;
      int qi = c >> 4, fc = (c & 15) * 8;  // 32 q-rows x 16 chunks of 8 ft
      bReg[i2] = *(const bf16x8*)(xbf + ((size_t)b * N_ + q0 + qi) * FT_ + n0 + fc);
    }
  };
  loadA(0);
  loadB(0);
  for (int q0 = 0; q0 < N_; q0 += BK) {
#pragma unroll
    for (int i2 = 0; i2 < 4; ++i2) {       // A^T: transpose-on-store -> [m][q]
      int c = tid + 128 * i2;
      int qi = c >> 4, mc = (c & 15) * 8;
#pragma unroll
      for (int e = 0; e < 8; ++e) As[(mc + e) * LDA + qi] = aReg[i2][e];
    }
#pragma unroll
    for (int i2 = 0; i2 < 4; ++i2) {       // B: transpose-on-store -> [ft][q]
      int c = tid + 128 * i2;
      int qi = c >> 4, fc = (c & 15) * 8;
#pragma unroll
      for (int e = 0; e < 8; ++e) Bs[(fc + e) * LDB + qi] = bReg[i2][e];
    }
    __syncthreads();
    if (q0 + BK < N_) {
      loadA(q0 + BK);
      loadB(q0 + BK);
      if (q0 + 2 * BK < N_) {
        __builtin_prefetch(cheb + ((size_t)kk * N_ + q0 + 2 * BK + (tid >> 4)) * N_ + m0, 0, 3);
        __builtin_prefetch(xbf + ((size_t)b * N_ + q0 + 2 * BK + (tid >> 4)) * FT_ + n0, 0, 3);
      }
    }
    mma_128x128(As, Bs, acc);
    __syncthreads();
  }
  const int lane = tid & 31, wave = tid >> 5;
  const int wm = (wave >> 1) & 1, wn = wave & 1, r = lane & 15, h = lane >> 4;
#pragma unroll
  for (int s0 = 0; s0 < 4; ++s0)
#pragma unroll
    for (int s1 = 0; s1 < 4; ++s1)
#pragma unroll
      for (int i = 0; i < 8; ++i) {
        int m = m0 + wm * 64 + s0 * 16 + 8 * h + i;
        int ft = n0 + wn * 64 + s1 * 16 + r;
        rbuf[(((size_t)kk * B_ + b) * N_ + m) * FT_ + ft] = acc[s0][s1][i];
      }
}

// ============================================================
// K11: fused epilogue per (b,m): Theta contraction + relu,
//      1x3 tconv + 1x1 rconv + relu, LayerNorm over TF, write out.
// ============================================================
__global__ __launch_bounds__(256) void k_fuse(const float* __restrict__ rbuf,
                                              const float* __restrict__ x,
                                              const float* __restrict__ Theta,
                                              const float* __restrict__ tcw,
                                              const float* __restrict__ tcb,
                                              const float* __restrict__ rcw,
                                              const float* __restrict__ rcb,
                                              const float* __restrict__ gamma,
                                              const float* __restrict__ beta,
                                              float* __restrict__ out) {
  const int bm = blockIdx.x;
  const int b = bm / N_, m = bm % N_;
  const int tid = threadIdx.x;
  __shared__ float rk[K_ * F_ * T_];
  __shared__ float xv[F_ * T_];
  __shared__ float gc[C_ * T_];
  __shared__ float yv[TF_ * T_];
  __shared__ float mu[T_], iv[T_];
  for (int e = tid; e < K_ * F_ * T_; e += 256) {
    int k = e / (F_ * T_), rm = e % (F_ * T_);
    rk[e] = rbuf[(((size_t)k * B_ + b) * N_ + m) * FT_ + rm];
  }
  for (int e = tid; e < F_ * T_; e += 256)
    xv[e] = x[((size_t)b * N_ + m) * FT_ + e];
  __syncthreads();
  for (int e = tid; e < C_ * T_; e += 256) {
    int c = e / T_, t = e % T_;
    float s = 0.f;
    for (int k = 0; k < K_; ++k)
      for (int f = 0; f < F_; ++f)
        s += rk[(k * F_ + f) * T_ + t] * Theta[(k * F_ + f) * C_ + c];
    gc[e] = fmaxf(s, 0.f);
  }
  __syncthreads();
  for (int e = tid; e < TF_ * T_; e += 256) {
    int tf = e / T_, t = e % T_;
    float s = tcb[tf] + rcb[tf];
    for (int c = 0; c < C_; ++c) {
      const float* wv = &tcw[((size_t)tf * C_ + c) * 3];
      if (t > 0)      s += gc[c * T_ + t - 1] * wv[0];
                      s += gc[c * T_ + t]     * wv[1];
      if (t < T_ - 1) s += gc[c * T_ + t + 1] * wv[2];
    }
    for (int f = 0; f < F_; ++f) s += xv[f * T_ + t] * rcw[tf * F_ + f];
    yv[e] = fmaxf(s, 0.f);
  }
  __syncthreads();
  if (tid < T_) {
    float s = 0.f, s2 = 0.f;
    for (int tf = 0; tf < TF_; ++tf) {
      float v = yv[tf * T_ + tid];
      s += v; s2 += v * v;
    }
    float mn = s / TF_;
    mu[tid] = mn;
    iv[tid] = rsqrtf(fmaxf(s2 / TF_ - mn * mn, 0.f) + LN_EPS);
  }
  __syncthreads();
  for (int e = tid; e < TF_ * T_; e += 256) {
    int tf = e / T_, t = e % T_;
    out[(((size_t)b * N_ + m) * TF_ + tf) * T_ + t] =
        (yv[e] - mu[t]) * iv[t] * gamma[tf] + beta[tf];
  }
}

// ============================================================
// launch
// ============================================================
extern "C" void kernel_launch(void* const* d_in, const int* in_sizes, int n_in,
                              void* d_out, int out_size, void* d_ws, size_t ws_size,
                              hipStream_t stream) {
  const float* x     = (const float*)d_in[0];
  const float* cheb  = (const float*)d_in[1];
  const float* U1    = (const float*)d_in[2];
  const float* U2    = (const float*)d_in[3];
  const float* U3    = (const float*)d_in[4];
  const float* be    = (const float*)d_in[5];
  const float* Ve    = (const float*)d_in[6];
  const float* W1    = (const float*)d_in[7];
  const float* W2    = (const float*)d_in[8];
  const float* W3    = (const float*)d_in[9];
  const float* bs    = (const float*)d_in[10];
  const float* Vs    = (const float*)d_in[11];
  const float* Theta = (const float*)d_in[12];
  const float* tcw   = (const float*)d_in[13];
  const float* tcb   = (const float*)d_in[14];
  const float* rcw   = (const float*)d_in[15];
  const float* rcb   = (const float*)d_in[16];
  const float* gam   = (const float*)d_in[17];
  const float* bet   = (const float*)d_in[18];
  float* out = (float*)d_out;

  char* w = (char*)d_ws;
  auto take = [&](size_t bytes) -> void* {
    void* p = (void*)w;
    w += (bytes + 255) & ~(size_t)255;
    return p;
  };
  float*  inner = (float*)take((size_t)B_ * T_ * F_ * 4);
  float*  rhsT  = (float*)take((size_t)B_ * N_ * T_ * 4);
  float*  tat   = (float*)take((size_t)B_ * T_ * T_ * 4);
  float*  x_tat = (float*)take((size_t)B_ * N_ * F_ * T_ * 4);
  __bf16* xbf   = (__bf16*)take((size_t)B_ * N_ * FT_ * 2);
  float*  lhs2  = (float*)take((size_t)B_ * N_ * T_ * 4);
  float*  rhs2v = (float*)take((size_t)B_ * N_ * T_ * 4);
  __bf16* Pbuf  = (__bf16*)take((size_t)B_ * N_ * N_ * 2);
  __bf16* Vsbf  = (__bf16*)take((size_t)N_ * N_ * 2);
  float*  S     = (float*)take((size_t)B_ * N_ * N_ * 4);
  __bf16* satb  = (__bf16*)take((size_t)B_ * N_ * N_ * 2);
  float*  rbuf  = (float*)take((size_t)K_ * B_ * N_ * FT_ * 4);

  // ---- temporal attention ----
  k_inner<<<dim3(B_ * F_ * T_), 256, 0, stream>>>(x, U1, inner);
  k_rhsT<<<dim3((B_ * N_ * T_ + 255) / 256), 256, 0, stream>>>(x, U3, rhsT);
  k_tat<<<dim3(B_), 256, 0, stream>>>(inner, rhsT, U2, Ve, be, tat);
  k_xtat<<<dim3((N_ * F_) / 256, B_), 256, 0, stream>>>(x, tat, x_tat);
  k_tobf<<<dim3((B_ * N_ * FT_ + 255) / 256), 256, 0, stream>>>(x, xbf, (size_t)B_ * N_ * FT_);

  // ---- spatial attention ----
  k_satprep<<<dim3((B_ * N_ + 255) / 256), 256, 0, stream>>>(x_tat, W1, W2, W3, lhs2, rhs2v);
  k_prod2<<<dim3(N_ / 256, N_, B_), 256, 0, stream>>>(lhs2, rhs2v, bs, Pbuf);
  k_tobf<<<dim3((N_ * N_ + 255) / 256), 256, 0, stream>>>(Vs, Vsbf, (size_t)N_ * N_);
  k_gemm1<<<dim3(N_ / BM, N_ / BN, B_), 128, 0, stream>>>(Vsbf, Pbuf, S);
  k_satsm<<<dim3(N_ / 64, B_), 64, 0, stream>>>(S, satb);

  // ---- gated Chebyshev graph conv (WMMA) ----
  k_gemm2<<<dim3(N_ / BM, FT_ / BN, B_ * K_), 128, 0, stream>>>(cheb, satb, xbf, rbuf);

  // ---- Theta + tconv + rconv + LN epilogue ----
  k_fuse<<<dim3(B_ * N_), 256, 0, stream>>>(rbuf, x, Theta, tcw, tcb, rcw, rcb,
                                            gam, bet, out);
}
